// MultiheadAttention_3642132267665
// MI455X (gfx1250) — compile-verified, matrix operand order checked by reading, not run on hardware
//
#include <hip/hip_runtime.h>
#include <hip/hip_bf16.h>

// ---------------------------------------------------------------------------
// MHA forward for MI455X (gfx1250, wave32, WMMA).
//   B=4, S=2048, E=1024, H=16, HD=64
// Outputs (concatenated in d_out): o [B,S,E] fp32, attn [B,H,S,S] fp32.
// All GEMMs run through v_wmma_f32_16x16x32_bf16 (f32 accumulate).
// Tile staging uses GLOBAL_LOAD_ASYNC_TO_LDS_B128 (ASYNCcnt) when available.
// ---------------------------------------------------------------------------

#define Bc 4
#define Sc 2048
#define Ec 1024
#define Hc 16
#define HDc 64

#if defined(__AMDGCN__) && __has_builtin(__builtin_amdgcn_global_load_async_to_lds_b128) && __has_builtin(__builtin_amdgcn_s_wait_asynccnt)
#define HAVE_ASYNC 1
#else
#define HAVE_ASYNC 0
#endif

typedef __attribute__((ext_vector_type(16))) __bf16 v16bf;  // 8 VGPRs
typedef __attribute__((ext_vector_type(8)))  float  v8f;    // 8 VGPRs
typedef __attribute__((ext_vector_type(4)))  int    v4i;    // 16 bytes

__device__ __forceinline__ __bf16 f2bf(float f) {
  unsigned u = __float_as_uint(f);
  u += 0x7FFFu + ((u >> 16) & 1u);            // round-to-nearest-even
  unsigned short h = (unsigned short)(u >> 16);
  __bf16 r;
  __builtin_memcpy(&r, &h, 2);
  return r;
}

// 16-byte global -> LDS copy.  Async path issues global_load_async_to_lds_b128
// (tracked by ASYNCcnt); fallback is a VGPR round-trip.
__device__ __forceinline__ void async_cp16(const __bf16* g, __bf16* l) {
#if HAVE_ASYNC
  __builtin_amdgcn_global_load_async_to_lds_b128(
      (__attribute__((address_space(1))) v4i*)g,
      (__attribute__((address_space(3))) v4i*)l, 0, 0);
#else
  *(uint4*)l = *(const uint4*)g;
#endif
}

__device__ __forceinline__ void async_wait() {
#if HAVE_ASYNC
  __builtin_amdgcn_s_wait_asynccnt(0);
#endif
}

// Load one 16x32 bf16 fragment (A layout; B uses the same packing on a
// pre-transposed [N][K] LDS tile).  Per CDNA5 ISA 16-bit A layout:
//   lanes 0-15 : row r, K = koff+0..7   and koff+16..23
//   lanes 16-31: row r, K = koff+8..15  and koff+24..31
__device__ __forceinline__ v16bf lds_frag(const __bf16* base, int row0,
                                          int stride, int koff) {
  int lane = threadIdx.x & 31;
  const __bf16* p = base + (row0 + (lane & 15)) * stride + koff + ((lane >> 4) << 3);
  union { v16bf v; uint4 q[2]; } u;
  u.q[0] = *(const uint4*)p;
  u.q[1] = *(const uint4*)(p + 16);
  return u.v;
}

__device__ __forceinline__ v8f wmma_bf16(v16bf a, v16bf b, v8f c) {
  return __builtin_amdgcn_wmma_f32_16x16x32_bf16(false, a, false, b,
                                                 (short)0, c, false, false);
}

// ---------------------------------------------------------------------------
// f32 -> bf16 convert (vectorized x4)
// ---------------------------------------------------------------------------
__global__ void cvt_f32_bf16(const float* __restrict__ src,
                             __bf16* __restrict__ dst, int n4) {
  int i = blockIdx.x * blockDim.x + threadIdx.x;
  if (i < n4) {
    float4 f = ((const float4*)src)[i];
    __bf16 t[4];
    t[0] = f2bf(f.x); t[1] = f2bf(f.y); t[2] = f2bf(f.z); t[3] = f2bf(f.w);
    *(uint2*)(dst + (size_t)i * 4) = *(uint2*)t;
  }
}

// ---------------------------------------------------------------------------
// Tiled bf16 WMMA GEMM:  C[M,N] = A[M,K] * B[K,N] + bias
//   block = 256 threads (8 waves), tile 128x128, K-step 32
//   waves laid out 4(M) x 2(N): each wave owns 32x64 -> 2x4 accumulators
//   Double-buffered LDS; A staged with async-to-LDS, B transposed via VGPRs.
// MODE 0: scatter into head-major bf16 Q/K/V buffers (QKV projection)
// MODE 1: fp32 row-major output (output projection)
// ---------------------------------------------------------------------------
#define ASTR 40   // LDS row stride (elements); 16B aligned, bank-spread
template <int MODE>
__global__ __launch_bounds__(256) void gemm_bf16_kernel(
    const __bf16* __restrict__ A, const __bf16* __restrict__ Bm,
    const float* __restrict__ bias, float* __restrict__ outF,
    __bf16* __restrict__ outQ, __bf16* __restrict__ outK,
    __bf16* __restrict__ outV, int M, int N, int K) {
  __shared__ __attribute__((aligned(16))) __bf16 As[2][128 * ASTR];
  __shared__ __attribute__((aligned(16))) __bf16 Bt[2][128 * ASTR];  // [n][k]

  const int tid  = threadIdx.x;
  const int lane = tid & 31;
  const int wave = tid >> 5;
  const int wm   = wave >> 1;   // 0..3
  const int wn   = wave & 1;    // 0..1
  const int m0   = blockIdx.y * 128;
  const int n0   = blockIdx.x * 128;

  v8f acc[2][4] = {};

  auto stageA = [&](int k0, int buf) {
    #pragma unroll
    for (int c = tid; c < 512; c += 256) {
      int r = c >> 2, cc = c & 3;
      async_cp16(A + (size_t)(m0 + r) * K + k0 + cc * 8,
                 &As[buf][r * ASTR + cc * 8]);
    }
  };
  auto stageB = [&](int k0, int buf) {
    #pragma unroll
    for (int c = tid; c < 512; c += 256) {
      int kr = c >> 4, cc = c & 15;
      uint4 v = *(const uint4*)(Bm + (size_t)(k0 + kr) * N + n0 + cc * 8);
      __bf16 t[8];
      *(uint4*)t = v;
      #pragma unroll
      for (int i = 0; i < 8; ++i) Bt[buf][(cc * 8 + i) * ASTR + kr] = t[i];
    }
  };

  stageA(0, 0);
  stageB(0, 0);

  for (int k0 = 0; k0 < K; k0 += 32) {
    const int buf = (k0 >> 5) & 1;
    async_wait();        // current tile's async copies done (own wave)
    __syncthreads();     // all waves' staging (async + ds stores) visible
    if (k0 + 32 < K) {   // kick off next tile into the other buffer
      stageA(k0 + 32, buf ^ 1);
      stageB(k0 + 32, buf ^ 1);
    }

    v16bf af[2], bfm[4];
    #pragma unroll
    for (int i = 0; i < 2; ++i)
      af[i] = lds_frag(As[buf], wm * 32 + i * 16, ASTR, 0);
    #pragma unroll
    for (int j = 0; j < 4; ++j)
      bfm[j] = lds_frag(Bt[buf], wn * 64 + j * 16, ASTR, 0);
    #pragma unroll
    for (int i = 0; i < 2; ++i)
      #pragma unroll
      for (int j = 0; j < 4; ++j)
        acc[i][j] = wmma_bf16(af[i], bfm[j], acc[i][j]);
  }

  // --- epilogue ---
  #pragma unroll
  for (int i = 0; i < 2; ++i) {
    #pragma unroll
    for (int j = 0; j < 4; ++j) {
      #pragma unroll
      for (int e = 0; e < 8; ++e) {
        int row = m0 + wm * 32 + i * 16 + ((lane >> 4) << 3) + e;
        int col = n0 + wn * 64 + j * 16 + (lane & 15);
        float v = acc[i][j][e] + bias[col];
        if (MODE == 1) {
          outF[(size_t)row * N + col] = v;
        } else {
          // col -> (head, which(q/k/v), dim); row -> (b, s)
          int h = col / (3 * HDc);
          int r = col % (3 * HDc);
          int which = r >> 6;
          int d = r & 63;
          int b = row >> 11;          // /S (2048)
          int s = row & (Sc - 1);
          size_t dst = (((size_t)(b * Hc + h) * Sc) + s) * HDc + d;
          __bf16 bv = f2bf(v);
          if (which == 0)      outQ[dst] = bv;
          else if (which == 1) outK[dst] = bv;
          else                 outV[dst] = bv;
        }
      }
    }
  }
}

// ---------------------------------------------------------------------------
// Fused causal attention per (b*H + h, 128-query block).
//   8 waves, each owns 16 query rows.  Two passes over key blocks of 64:
//   pass 1 -> online softmax stats, pass 2 -> write probs + PV via WMMA.
//   Q and K tiles staged with async-to-LDS; V transposed via VGPR path.
// ---------------------------------------------------------------------------
#define PSTR 72   // padded LDS stride for 64-wide tiles
__global__ __launch_bounds__(256) void attn_kernel(
    const __bf16* __restrict__ Q, const __bf16* __restrict__ Kb,
    const __bf16* __restrict__ Vb, float* __restrict__ attn_out,
    __bf16* __restrict__ vals) {
  __shared__ __attribute__((aligned(16))) __bf16 Qs[128 * PSTR];
  __shared__ __attribute__((aligned(16))) __bf16 Ks[64 * PSTR];
  __shared__ __attribute__((aligned(16))) __bf16 Vt[64 * PSTR];   // [d][key]
  __shared__ __attribute__((aligned(16))) __bf16 Ps[128 * PSTR];

  const int tid  = threadIdx.x;
  const int lane = tid & 31;
  const int wv   = tid >> 5;            // wave id 0..7
  const int bh   = blockIdx.y;          // b*H + h
  const int q0   = blockIdx.x * 128;
  const float scale = 0.125f;           // 1/sqrt(64)

  const __bf16* Qh = Q  + (size_t)bh * Sc * HDc;
  const __bf16* Kh = Kb + (size_t)bh * Sc * HDc;
  const __bf16* Vh = Vb + (size_t)bh * Sc * HDc;

  // stage Q tile 128x64 once (async)
  #pragma unroll
  for (int c = tid; c < 1024; c += 256) {
    int r = c >> 3, cc = c & 7;
    async_cp16(Qh + (size_t)(q0 + r) * HDc + cc * 8, Qs + r * PSTR + cc * 8);
  }

  float m8[8], l8[8];
  #pragma unroll
  for (int e = 0; e < 8; ++e) { m8[e] = -3.0e38f; l8[e] = 0.f; }

  const int kend = q0 + 128;            // exclusive causal bound

  // ---------------- pass 1: softmax statistics ----------------
  for (int j0 = 0; j0 < kend; j0 += 64) {
    __syncthreads();
    #pragma unroll
    for (int c = tid; c < 512; c += 256) {
      int r = c >> 3, cc = c & 7;
      async_cp16(Kh + (size_t)(j0 + r) * HDc + cc * 8, Ks + r * PSTR + cc * 8);
    }
    async_wait();
    __syncthreads();

    v16bf a0 = lds_frag(Qs, wv * 16, PSTR, 0);
    v16bf a1 = lds_frag(Qs, wv * 16, PSTR, 32);
    v8f xs[4];
    #pragma unroll
    for (int nf = 0; nf < 4; ++nf) {
      v16bf b0 = lds_frag(Ks, nf * 16, PSTR, 0);
      v16bf b1 = lds_frag(Ks, nf * 16, PSTR, 32);
      v8f a = {};
      a = wmma_bf16(a0, b0, a);
      a = wmma_bf16(a1, b1, a);
      xs[nf] = a;
    }
    // mask + scale, then row-wise online max/sum
    #pragma unroll
    for (int e = 0; e < 8; ++e) {
      int qg = q0 + wv * 16 + ((lane >> 4) << 3) + e;
      float bm = -3.0e38f;
      #pragma unroll
      for (int nf = 0; nf < 4; ++nf) {
        int kg = j0 + nf * 16 + (lane & 15);
        float x = (kg <= qg) ? xs[nf][e] * scale : -3.0e38f;
        xs[nf][e] = x;
        bm = fmaxf(bm, x);
      }
      bm = fmaxf(bm, __shfl_xor(bm, 1, 32));
      bm = fmaxf(bm, __shfl_xor(bm, 2, 32));
      bm = fmaxf(bm, __shfl_xor(bm, 4, 32));
      bm = fmaxf(bm, __shfl_xor(bm, 8, 32));
      float nm = fmaxf(m8[e], bm);
      float s = 0.f;
      #pragma unroll
      for (int nf = 0; nf < 4; ++nf) {
        float x = xs[nf][e];
        s += (x > -1.0e37f) ? __expf(x - nm) : 0.f;
      }
      s += __shfl_xor(s, 1, 32);
      s += __shfl_xor(s, 2, 32);
      s += __shfl_xor(s, 4, 32);
      s += __shfl_xor(s, 8, 32);
      l8[e] = l8[e] * __expf(m8[e] - nm) + s;
      m8[e] = nm;
    }
  }

  float rl[8];
  #pragma unroll
  for (int e = 0; e < 8; ++e) rl[e] = (l8[e] > 0.f) ? (1.f / l8[e]) : 0.f;

  // ---------------- pass 2: write probs + PV ----------------
  v8f oacc[4] = {};
  for (int j0 = 0; j0 < kend; j0 += 64) {
    __syncthreads();
    #pragma unroll
    for (int c = tid; c < 512; c += 256) {
      int r = c >> 3, cc = c & 7;
      async_cp16(Kh + (size_t)(j0 + r) * HDc + cc * 8, Ks + r * PSTR + cc * 8);
      // V staged transposed: Vt[d][key] (VGPR path)
      uint4 v = *(const uint4*)(Vh + (size_t)(j0 + r) * HDc + cc * 8);
      __bf16 t[8];
      *(uint4*)t = v;
      #pragma unroll
      for (int i = 0; i < 8; ++i) Vt[(cc * 8 + i) * PSTR + r] = t[i];
    }
    async_wait();
    __syncthreads();

    v16bf a0 = lds_frag(Qs, wv * 16, PSTR, 0);
    v16bf a1 = lds_frag(Qs, wv * 16, PSTR, 32);
    v8f xs[4];
    #pragma unroll
    for (int nf = 0; nf < 4; ++nf) {
      v16bf b0 = lds_frag(Ks, nf * 16, PSTR, 0);
      v16bf b1 = lds_frag(Ks, nf * 16, PSTR, 32);
      v8f a = {};
      a = wmma_bf16(a0, b0, a);
      a = wmma_bf16(a1, b1, a);
      xs[nf] = a;
    }
    #pragma unroll
    for (int nf = 0; nf < 4; ++nf) {
      #pragma unroll
      for (int e = 0; e < 8; ++e) {
        int rloc = wv * 16 + ((lane >> 4) << 3) + e;
        int qg = q0 + rloc;
        int kg = j0 + nf * 16 + (lane & 15);
        float x = (kg <= qg) ? xs[nf][e] * scale : -3.0e38f;
        float p = (x > -1.0e37f) ? __expf(x - m8[e]) * rl[e] : 0.f;
        attn_out[((size_t)bh * Sc + qg) * Sc + kg] = p;
        Ps[rloc * PSTR + nf * 16 + (lane & 15)] = f2bf(p);
      }
    }
    __syncthreads();   // Ps visible (and uniform sync before next K/V stage)

    v16bf p0 = lds_frag(Ps, wv * 16, PSTR, 0);
    v16bf p1 = lds_frag(Ps, wv * 16, PSTR, 32);
    #pragma unroll
    for (int nf = 0; nf < 4; ++nf) {
      v16bf vb0 = lds_frag(Vt, nf * 16, PSTR, 0);
      v16bf vb1 = lds_frag(Vt, nf * 16, PSTR, 32);
      oacc[nf] = wmma_bf16(p0, vb0, oacc[nf]);
      oacc[nf] = wmma_bf16(p1, vb1, oacc[nf]);
    }
  }

  // write head outputs back to [B,S,E] (embed-major) bf16 for the out-proj
  int b = bh >> 4, h = bh & 15;
  #pragma unroll
  for (int nf = 0; nf < 4; ++nf) {
    #pragma unroll
    for (int e = 0; e < 8; ++e) {
      int s = q0 + wv * 16 + ((lane >> 4) << 3) + e;
      int d = nf * 16 + (lane & 15);
      vals[((size_t)(b * Sc + s)) * Ec + h * HDc + d] = f2bf(oacc[nf][e]);
    }
  }
}

// ---------------------------------------------------------------------------
// Launcher.  Inputs: x, w_qkv, b_qkv, w_o, b_o, mask (mask unused; causal
// mask computed analytically).  d_out = [o (B*S*E) | attn (B*H*S*S)] fp32.
// ---------------------------------------------------------------------------
extern "C" void kernel_launch(void* const* d_in, const int* in_sizes, int n_in,
                              void* d_out, int out_size, void* d_ws,
                              size_t ws_size, hipStream_t stream) {
  (void)in_sizes; (void)n_in; (void)out_size; (void)ws_size;
  const float* x    = (const float*)d_in[0];
  const float* wqkv = (const float*)d_in[1];
  const float* bqkv = (const float*)d_in[2];
  const float* wo   = (const float*)d_in[3];
  const float* bo   = (const float*)d_in[4];

  const size_t nX  = (size_t)Bc * Sc * Ec;        // 8,388,608
  const size_t nWq = (size_t)Ec * 3 * Ec;         // 3,145,728
  const size_t nWo = (size_t)Ec * Ec;             // 1,048,576
  const size_t nHd = (size_t)Bc * Hc * Sc * HDc;  // 8,388,608

  char* ws = (char*)d_ws;
  size_t off = 0;
  auto take = [&](size_t bytes) {
    void* p = ws + off;
    off += (bytes + 255) & ~(size_t)255;
    return p;
  };
  __bf16* Xbf   = (__bf16*)take(nX * 2);
  __bf16* Wqkvb = (__bf16*)take(nWq * 2);
  __bf16* Wob   = (__bf16*)take(nWo * 2);
  __bf16* Qb    = (__bf16*)take(nHd * 2);
  __bf16* Kb    = (__bf16*)take(nHd * 2);
  __bf16* Vb    = (__bf16*)take(nHd * 2);
  __bf16* Valsb = (__bf16*)take(nX * 2);

  // 1) converts
  {
    int n4 = (int)(nX / 4);
    cvt_f32_bf16<<<(n4 + 255) / 256, 256, 0, stream>>>(x, Xbf, n4);
    n4 = (int)(nWq / 4);
    cvt_f32_bf16<<<(n4 + 255) / 256, 256, 0, stream>>>(wqkv, Wqkvb, n4);
    n4 = (int)(nWo / 4);
    cvt_f32_bf16<<<(n4 + 255) / 256, 256, 0, stream>>>(wo, Wob, n4);
  }

  // 2) fused QKV projection -> head-major bf16 Q/K/V
  {
    dim3 grid(3 * Ec / 128, (Bc * Sc) / 128);  // (24, 64)
    gemm_bf16_kernel<0><<<grid, 256, 0, stream>>>(
        Xbf, Wqkvb, bqkv, nullptr, Qb, Kb, Vb, Bc * Sc, 3 * Ec, Ec);
  }

  // 3) zero attn region (upper triangle is never written by attn_kernel)
  float* attn_out = (float*)d_out + nX;
  (void)hipMemsetAsync(attn_out, 0,
                       (size_t)Bc * Hc * Sc * Sc * sizeof(float), stream);

  // 4) fused causal attention (writes attn probs + bf16 head outputs)
  {
    dim3 grid(Sc / 128, Bc * Hc);  // (16, 64)
    attn_kernel<<<grid, 256, 0, stream>>>(Qb, Kb, Vb, attn_out, Valsb);
  }

  // 5) output projection -> fp32 o
  {
    dim3 grid(Ec / 128, (Bc * Sc) / 128);  // (8, 64)
    gemm_bf16_kernel<1><<<grid, 256, 0, stream>>>(
        Valsb, Wob, bo, (float*)d_out, nullptr, nullptr, nullptr,
        Bc * Sc, Ec, Ec);
  }
}
